// Qwen3NextAttentionForEngine_30872224924282
// MI455X (gfx1250) — matrix operand order repaired and checked by reading, not compiled, not run on hardware
//
#include <hip/hip_runtime.h>
#include <hip/hip_bf16.h>

#define HIDN   2048
#define NH     16
#define NKV    4
#define HD     128
#define SEQ    2048
#define BATCH  2
#define TTOK   (BATCH * SEQ)
#define QGN    (2 * HIDN)          // q + gate projection width
#define KVN    (NKV * HD)          // 512
#define SCALE  0.088388347648318447f  // 1/sqrt(128), folded into q pack
#define NEGBIG (-1.0e9f)

typedef __attribute__((ext_vector_type(8)))  float  v8f;
typedef __attribute__((ext_vector_type(8)))  __bf16 v8bf;
typedef __attribute__((ext_vector_type(16))) __bf16 v16bf;

static __device__ __forceinline__ v8f wmma_bf16(v16bf a, v16bf b, v8f c) {
  return __builtin_amdgcn_wmma_f32_16x16x32_bf16(false, a, false, b, (short)0, c,
                                                 false, false);
}

// Async DMA global->LDS (CDNA5, ASYNCcnt-tracked, bypasses VGPRs).
static __device__ __forceinline__ void async_copy16(unsigned lds_off,
                                                    unsigned long long gaddr) {
  asm volatile("global_load_async_to_lds_b128 %0, %1, off"
               :: "v"(lds_off), "v"(gaddr) : "memory");
}
static __device__ __forceinline__ void wait_async0() {
  asm volatile("s_wait_asynccnt 0x0" ::: "memory");
}
static __device__ __forceinline__ v16bf comb16(v8bf a, v8bf b) {
  v16bf r;
#pragma unroll
  for (int e = 0; e < 8; ++e) { r[e] = a[e]; r[8 + e] = b[e]; }
  return r;
}

// ---------------------------------------------------------------------------
// Kernel 0: split f32 -> bf16 (hi, lo) pair:  x = hi + lo, |lo| <= ulp(hi)/2
// ---------------------------------------------------------------------------
__global__ __launch_bounds__(256) void split_bf16_kernel(
    const float* __restrict__ x, __bf16* __restrict__ hi,
    __bf16* __restrict__ lo, int n) {
  const int i = (blockIdx.x * 256 + threadIdx.x) * 4;
  if (i >= n) return;
#pragma unroll
  for (int e = 0; e < 4; ++e) {
    const float v = x[i + e];
    const __bf16 h = (__bf16)v;
    hi[i + e] = h;
    lo[i + e] = (__bf16)(v - (float)h);
  }
}

// ---------------------------------------------------------------------------
// Kernel 1: bf16x3 split-precision GEMM  Y[M,N] = X[M,K] * W[N,K]^T  (f32 acc)
//   X = Xh + Xl, W = Wh + Wl;  Y ~= Xh*Wh + Xh*Wl + Xl*Wh   (lo*lo dropped)
// 128 threads / 4 waves, tile 64x64, BK=32.  Async-to-LDS double buffer,
// XOR-swizzled 16B groups so DMA stores are contiguous and ds reads are
// bank-conflict-free.  12 bf16 WMMAs per K-chunk per wave.
// ---------------------------------------------------------------------------
__global__ __launch_bounds__(128) void gemm_bf16x3_kernel(
    const __bf16* __restrict__ Xh, const __bf16* __restrict__ Xl,
    const __bf16* __restrict__ Wh, const __bf16* __restrict__ Wl,
    float* __restrict__ Y, int M, int N, int K) {
  __shared__ __bf16 sm[2][4][64][32];   // [buf][Xh,Xl,Wh,Wl][row][k-half]

  const int tid  = threadIdx.x;
  const int wave = tid >> 5;
  const int lane = tid & 31;
  const int lr   = lane & 15;
  const int lh   = lane >> 4;
  const int m0   = blockIdx.x * 64;
  const int n0   = blockIdx.y * 64;
  const int wm   = (wave >> 1) * 32;
  const int wn   = (wave & 1) * 32;

  const __bf16* srcs[4] = { Xh + (size_t)m0 * K, Xl + (size_t)m0 * K,
                            Wh + (size_t)n0 * K, Wl + (size_t)n0 * K };

  v8f acc[2][2] = {};

  // DMA one 64x32-half tile per array into buffer `buf` (16B per lane-slot).
  auto issue = [&](int buf, int k0) {
#pragma unroll
    for (int arr = 0; arr < 4; ++arr) {
#pragma unroll
      for (int i = 0; i < 2; ++i) {
        const int v   = tid * 2 + i;           // 256 x 16B slots per array
        const int row = v >> 2;
        const int g   = v & 3;                 // logical 8-half group
        const int pg  = g ^ ((row >> 2) & 3);  // swizzled group
        const unsigned l = (unsigned)(size_t)&sm[buf][arr][row][pg * 8];
        const unsigned long long ga =
            (unsigned long long)(size_t)(srcs[arr] + (size_t)row * K + k0 + g * 8);
        async_copy16(l, ga);
      }
    }
  };

  // A frag (16x32): lane holds row lr, K = b0..b0+15 (contiguous), b0=16*lh
  auto loadA = [&](int buf, int arr, int mt) -> v16bf {
    const int row = wm + mt * 16 + lr;
    const int s   = (row >> 2) & 3;
    const int g0  = 2 * lh;                    // b0/8
    const v8bf a = *(const v8bf*)&sm[buf][arr][row][((g0)     ^ s) * 8];
    const v8bf b = *(const v8bf*)&sm[buf][arr][row][((g0 + 1) ^ s) * 8];
    return comb16(a, b);
  };
  // B frag (32x16): lane holds col lr, K = lh*16 .. lh*16+15 (contiguous)
  auto loadB = [&](int buf, int arr, int nt) -> v16bf {
    const int row = wn + nt * 16 + lr;
    const int s   = (row >> 2) & 3;
    const v8bf a = *(const v8bf*)&sm[buf][arr][row][((2 * lh)     ^ s) * 8];
    const v8bf b = *(const v8bf*)&sm[buf][arr][row][((2 * lh + 1) ^ s) * 8];
    return comb16(a, b);
  };

  issue(0, 0);
  for (int k0 = 0; k0 < K; k0 += 32) {
    const int buf = (k0 >> 5) & 1;
    wait_async0();        // my DMA for `buf` done
    __syncthreads();      // everyone's DMA done + prior reads of buf^1 retired
    if (k0 + 32 < K) issue(buf ^ 1, k0 + 32);  // overlap next DMA with WMMAs

    const v16bf Ah0 = loadA(buf, 0, 0), Ah1 = loadA(buf, 0, 1);
    const v16bf Al0 = loadA(buf, 1, 0), Al1 = loadA(buf, 1, 1);
    const v16bf Bh0 = loadB(buf, 2, 0), Bh1 = loadB(buf, 2, 1);
    const v16bf Bl0 = loadB(buf, 3, 0), Bl1 = loadB(buf, 3, 1);

    acc[0][0] = wmma_bf16(Ah0, Bh0, acc[0][0]);
    acc[0][1] = wmma_bf16(Ah0, Bh1, acc[0][1]);
    acc[1][0] = wmma_bf16(Ah1, Bh0, acc[1][0]);
    acc[1][1] = wmma_bf16(Ah1, Bh1, acc[1][1]);
    acc[0][0] = wmma_bf16(Ah0, Bl0, acc[0][0]);
    acc[0][1] = wmma_bf16(Ah0, Bl1, acc[0][1]);
    acc[1][0] = wmma_bf16(Ah1, Bl0, acc[1][0]);
    acc[1][1] = wmma_bf16(Ah1, Bl1, acc[1][1]);
    acc[0][0] = wmma_bf16(Al0, Bh0, acc[0][0]);
    acc[0][1] = wmma_bf16(Al0, Bh1, acc[0][1]);
    acc[1][0] = wmma_bf16(Al1, Bh0, acc[1][0]);
    acc[1][1] = wmma_bf16(Al1, Bh1, acc[1][1]);
  }

  // C layout: VGPR e -> row (e + 8*lh), col lr
#pragma unroll
  for (int i = 0; i < 2; ++i)
#pragma unroll
    for (int j = 0; j < 2; ++j)
#pragma unroll
      for (int e = 0; e < 8; ++e) {
        const int row = m0 + wm + i * 16 + e + 8 * lh;
        const int col = n0 + wn + j * 16 + lr;
        Y[(size_t)row * N + col] = acc[i][j][e];
      }
}

// ---------------------------------------------------------------------------
// Kernel 2: per-head RMS-norm + RoPE; pack q (pre-scaled by 1/sqrt(HD)) and k
// to bf16 [B,H,S,D]; pack v to bf16 TRANSPOSED [B,KV,D,S] so attention B-frags
// are single contiguous 32B loads.
// ---------------------------------------------------------------------------
__global__ __launch_bounds__(256) void normrope_kernel(
    const float* __restrict__ qg, const float* __restrict__ kf,
    const float* __restrict__ vf, const float* __restrict__ cosb,
    const float* __restrict__ sinb, const float* __restrict__ qw,
    const float* __restrict__ kw, __bf16* __restrict__ qb,
    __bf16* __restrict__ kb, __bf16* __restrict__ vb) {
  const int t    = blockIdx.x;
  const int b    = t / SEQ;
  const int s    = t % SEQ;
  const int wave = threadIdx.x >> 5;
  const int lane = threadIdx.x & 31;

  for (int job = wave; job < NH + 2 * NKV; job += 8) {
    if (job < NH + NKV) {
      const bool isq = job < NH;
      const int h    = isq ? job : job - NH;
      const float* x = isq ? (qg + (size_t)t * QGN + h * HD)
                           : (kf + (size_t)t * KVN + h * HD);
      const float* w = isq ? qw : kw;
      float xv[4];
      float ss = 0.f;
#pragma unroll
      for (int e = 0; e < 4; ++e) {
        xv[e] = x[lane * 4 + e];
        ss += xv[e] * xv[e];
      }
#pragma unroll
      for (int off = 16; off >= 1; off >>= 1) ss += __shfl_xor(ss, off, 32);
      const float r = rsqrtf(ss * (1.0f / HD) + 1e-6f);
#pragma unroll
      for (int e = 0; e < 4; ++e) {
        const int i  = lane * 4 + e;
        const int ip = i ^ 64;                   // rotate_half partner
        const float nv = w[i] * xv[e] * r;
        const float np = w[ip] * x[ip] * r;
        const float rot = (i < 64) ? -np : np;
        float o = nv * cosb[s * HD + i] + rot * sinb[s * HD + i];
        if (isq) {
          o *= SCALE;  // fold softmax scale into q
          qb[((size_t)(b * NH + h) * SEQ + s) * HD + i] = (__bf16)o;
        } else {
          kb[((size_t)(b * NKV + h) * SEQ + s) * HD + i] = (__bf16)o;
        }
      }
    } else {
      const int h = job - NH - NKV;
#pragma unroll
      for (int e = 0; e < 4; ++e) {
        const int i = lane * 4 + e;
        vb[((size_t)(b * NKV + h) * HD + i) * SEQ + s] =
            (__bf16)vf[(size_t)t * KVN + h * HD + i];
      }
    }
  }
}

// ---------------------------------------------------------------------------
// Kernel 3: causal flash attention, bf16 WMMA, online softmax, sigmoid gate.
// Grid (SEQ/128, NH, B); 8 waves; each wave owns 16 query rows.
// Gated output written directly as bf16 hi/lo for the bf16x3 Wo projection.
// ---------------------------------------------------------------------------
__global__ __launch_bounds__(256) void flash_attn_kernel(
    const __bf16* __restrict__ qb, const __bf16* __restrict__ kbuf,
    const __bf16* __restrict__ vbuf, const float* __restrict__ qg,
    __bf16* __restrict__ attnh, __bf16* __restrict__ attnl) {
  const int h     = blockIdx.y;
  const int b     = blockIdx.z;
  const int wave  = threadIdx.x >> 5;
  const int lane  = threadIdx.x & 31;
  const int lr    = lane & 15;
  const int lh    = lane >> 4;
  const int qbase = blockIdx.x * 128 + wave * 16;
  const int kvh   = h / (NH / NKV);

  const __bf16* Q  = qb   + ((size_t)(b * NH + h) * SEQ + qbase) * HD;
  const __bf16* Kp = kbuf + (size_t)(b * NKV + kvh) * SEQ * HD;
  const __bf16* Vp = vbuf + (size_t)(b * NKV + kvh) * HD * SEQ;  // [D][S]

  __shared__ __bf16 plds[8][16][40];   // per-wave prob tile, padded stride

  const int b0 = lh * 16;
  v16bf qf[4];
#pragma unroll
  for (int c = 0; c < 4; ++c) {
    const __bf16* qr = Q + (size_t)lr * HD + c * 32;
    qf[c] = comb16(*(const v8bf*)(qr + b0), *(const v8bf*)(qr + b0 + 8));
  }

  v8f o[8] = {};
  float mrow[8], lrow[8];
#pragma unroll
  for (int j = 0; j < 8; ++j) { mrow[j] = -1.0e30f; lrow[j] = 0.f; }

  const int kend = qbase + 16;
  for (int kb0 = 0; kb0 < kend; kb0 += 32) {
    // ---- scores: two 16x16 tiles ----
    v8f s0 = {}, s1 = {};
#pragma unroll
    for (int c = 0; c < 4; ++c) {
      const v16bf bf0 = *(const v16bf*)(Kp + (size_t)(kb0 + lr) * HD + c * 32 + lh * 16);
      const v16bf bf1 = *(const v16bf*)(Kp + (size_t)(kb0 + 16 + lr) * HD + c * 32 + lh * 16);
      s0 = wmma_bf16(qf[c], bf0, s0);
      s1 = wmma_bf16(qf[c], bf1, s1);
    }

    // ---- online softmax ----
#pragma unroll
    for (int j = 0; j < 8; ++j) {
      const int qrow = qbase + j + 8 * lh;
      float e0 = s0[j];                 // scale already folded into q
      float e1 = s1[j];
      if (kb0 + lr > qrow)      e0 = NEGBIG;
      if (kb0 + 16 + lr > qrow) e1 = NEGBIG;
      float tm = fmaxf(e0, e1);
#pragma unroll
      for (int off = 8; off >= 1; off >>= 1) tm = fmaxf(tm, __shfl_xor(tm, off, 32));
      const float nm = fmaxf(mrow[j], tm);
      const float p0 = __expf(e0 - nm);
      const float p1 = __expf(e1 - nm);
      float rs = p0 + p1;
#pragma unroll
      for (int off = 8; off >= 1; off >>= 1) rs += __shfl_xor(rs, off, 32);
      const float corr = __expf(mrow[j] - nm);
      lrow[j] = lrow[j] * corr + rs;
      mrow[j] = nm;
#pragma unroll
      for (int ht = 0; ht < 8; ++ht) o[ht][j] *= corr;
      plds[wave][j + 8 * lh][lr]      = (__bf16)p0;
      plds[wave][j + 8 * lh][16 + lr] = (__bf16)p1;
    }

    // wave-private LDS round trip: C-layout probs -> A-layout fragment
    asm volatile("s_wait_dscnt 0x0" ::: "memory");
    v16bf pf;
    {
      const __bf16* pr = &plds[wave][lr][0];
      pf = comb16(*(const v8bf*)(pr + b0), *(const v8bf*)(pr + b0 + 8));
    }

    // ---- PV: V is [D][S], so each B-frag column is one contiguous 32B load
#pragma unroll
    for (int ht = 0; ht < 8; ++ht) {
      const v16bf vfrag =
          *(const v16bf*)(Vp + (size_t)(ht * 16 + lr) * SEQ + kb0 + lh * 16);
      o[ht] = wmma_bf16(pf, vfrag, o[ht]);
    }
  }

  // ---- epilogue: normalize, sigmoid gate, split-write bf16 hi/lo ----
#pragma unroll
  for (int ht = 0; ht < 8; ++ht)
#pragma unroll
    for (int j = 0; j < 8; ++j) {
      const int qrow = qbase + j + 8 * lh;
      const size_t tt = (size_t)b * SEQ + qrow;
      const int col = h * HD + ht * 16 + lr;
      const float g = qg[tt * QGN + HIDN + col];
      const float val = (o[ht][j] / lrow[j]) * (1.f / (1.f + __expf(-g)));
      const __bf16 hv = (__bf16)val;
      attnh[tt * HIDN + col] = hv;
      attnl[tt * HIDN + col] = (__bf16)(val - (float)hv);
    }
}

// ---------------------------------------------------------------------------
extern "C" void kernel_launch(void* const* d_in, const int* in_sizes, int n_in,
                              void* d_out, int out_size, void* d_ws, size_t ws_size,
                              hipStream_t stream) {
  const float* hidden = (const float*)d_in[0];
  // d_in[1] = attention_mask: causal mask computed inline
  const float* cosb = (const float*)d_in[2];
  const float* sinb = (const float*)d_in[3];
  const float* Wq   = (const float*)d_in[4];
  const float* Wk   = (const float*)d_in[5];
  const float* Wv   = (const float*)d_in[6];
  const float* Wo   = (const float*)d_in[7];
  const float* qw   = (const float*)d_in[8];
  const float* kw   = (const float*)d_in[9];
  float* out = (float*)d_out;

  char* ws = (char*)d_ws;
  auto alloc_f32  = [&](size_t n) { float*  p = (float*)ws;  ws += n * sizeof(float);  return p; };
  auto alloc_bf16 = [&](size_t n) { __bf16* p = (__bf16*)ws; ws += n * sizeof(__bf16); return p; };

  float*  qg   = alloc_f32((size_t)TTOK * QGN);
  float*  kf   = alloc_f32((size_t)TTOK * KVN);
  float*  vf   = alloc_f32((size_t)TTOK * KVN);
  __bf16* qb   = alloc_bf16((size_t)TTOK * HIDN);
  __bf16* kb   = alloc_bf16((size_t)TTOK * KVN);
  __bf16* vb   = alloc_bf16((size_t)TTOK * KVN);
  __bf16* Xh   = alloc_bf16((size_t)TTOK * HIDN);
  __bf16* Xl   = alloc_bf16((size_t)TTOK * HIDN);
  __bf16* Wqh  = alloc_bf16((size_t)QGN * HIDN);
  __bf16* Wql  = alloc_bf16((size_t)QGN * HIDN);
  __bf16* Wkh  = alloc_bf16((size_t)KVN * HIDN);
  __bf16* Wkl  = alloc_bf16((size_t)KVN * HIDN);
  __bf16* Wvh  = alloc_bf16((size_t)KVN * HIDN);
  __bf16* Wvl  = alloc_bf16((size_t)KVN * HIDN);
  __bf16* Woh  = alloc_bf16((size_t)HIDN * HIDN);
  __bf16* Wol  = alloc_bf16((size_t)HIDN * HIDN);
  __bf16* ath  = alloc_bf16((size_t)TTOK * HIDN);
  __bf16* atl  = alloc_bf16((size_t)TTOK * HIDN);

  // ---- split-precision decompositions (activations + weights) ----
  auto split = [&](const float* x, __bf16* h, __bf16* l, int n) {
    split_bf16_kernel<<<dim3(n / 1024), dim3(256), 0, stream>>>(x, h, l, n);
  };
  split(hidden, Xh, Xl, TTOK * HIDN);
  split(Wq, Wqh, Wql, QGN * HIDN);
  split(Wk, Wkh, Wkl, KVN * HIDN);
  split(Wv, Wvh, Wvl, KVN * HIDN);
  split(Wo, Woh, Wol, HIDN * HIDN);

  // ---- projections (bf16x3, async-DMA double-buffered WMMA GEMM) ----
  const dim3 gblk(128);
  gemm_bf16x3_kernel<<<dim3(TTOK / 64, QGN / 64), gblk, 0, stream>>>(
      Xh, Xl, Wqh, Wql, qg, TTOK, QGN, HIDN);
  gemm_bf16x3_kernel<<<dim3(TTOK / 64, KVN / 64), gblk, 0, stream>>>(
      Xh, Xl, Wkh, Wkl, kf, TTOK, KVN, HIDN);
  gemm_bf16x3_kernel<<<dim3(TTOK / 64, KVN / 64), gblk, 0, stream>>>(
      Xh, Xl, Wvh, Wvl, vf, TTOK, KVN, HIDN);

  normrope_kernel<<<dim3(TTOK), dim3(256), 0, stream>>>(
      qg, kf, vf, cosb, sinb, qw, kw, qb, kb, vb);

  flash_attn_kernel<<<dim3(SEQ / 128, NH, BATCH), dim3(256), 0, stream>>>(
      qb, kb, vb, qg, ath, atl);

  gemm_bf16x3_kernel<<<dim3(TTOK / 64, HIDN / 64), gblk, 0, stream>>>(
      ath, atl, Woh, Wol, out, TTOK, HIDN, HIDN);
}
